// LSTMNet_83880711291258
// MI455X (gfx1250) — matrix operand order repaired and checked by reading, not compiled, not run on hardware
//
#include <hip/hip_runtime.h>
#include <math.h>

// ---- problem constants (match reference) ----
#define VV 32000
#define HH 1024
#define BB 64
#define LL 2
#define N4H (4 * HH)   // 4096
#define INV_TEMP 1.25f // 1/0.8

typedef __attribute__((ext_vector_type(16))) __bf16 bf16x16;
typedef __attribute__((ext_vector_type(8)))  float  floatx8;
typedef __attribute__((ext_vector_type(4)))  float  floatx4;

// Load two runs of 8 fp32 (p[0..7], p[16..23]) and pack to bf16 (B-fragment per-lane layout:
// element e<8 -> K = (half?8:0)+e, e>=8 -> K = 16+(half?8:0)+(e-8), column N = lane&15).
__device__ __forceinline__ bf16x16 cvt_load16(const float* __restrict__ p) {
    const floatx4* q = (const floatx4*)p;
    floatx4 f0 = q[0], f1 = q[1], f2 = q[4], f3 = q[5];
    bf16x16 r;
#pragma unroll
    for (int i = 0; i < 4; ++i) {
        r[i]      = (__bf16)f0[i];
        r[4 + i]  = (__bf16)f1[i];
        r[8 + i]  = (__bf16)f2[i];
        r[12 + i] = (__bf16)f3[i];
    }
    return r;
}

// A-fragment (16x32 bf16): lanes 0-15 hold K runs [0..7] and [16..23];
// lanes 16-31 hold [8..15] and [24..31]  (ISA 05_wmma 16-bit A layout).
__device__ __forceinline__ bf16x16 cvt_loadA(const float* __restrict__ row, int k0, int half) {
    const float* p0 = row + k0 + (half ? 8 : 0);
    const floatx4* q0 = (const floatx4*)p0;
    const floatx4* q1 = (const floatx4*)(p0 + 16);
    floatx4 f0 = q0[0], f1 = q0[1], f2 = q1[0], f3 = q1[1];
    bf16x16 r;
#pragma unroll
    for (int i = 0; i < 4; ++i) {
        r[i]      = (__bf16)f0[i];
        r[4 + i]  = (__bf16)f1[i];
        r[8 + i]  = (__bf16)f2[i];
        r[12 + i] = (__bf16)f3[i];
    }
    return r;
}

// Partial GEMM: out_partial[slice][n][b] += A[b,:]·W[n,:] over this block's K-chunk.
// A: [64, K] row-major.  W: [N, K] row-major.  grid = (N/128, S), block = 256 (8 waves).
// Each wave owns one 16-column N tile and all 4 M(=batch) tiles.
__global__ void __launch_bounds__(256)
gemm_partial(const float* __restrict__ A, const float* __restrict__ W,
             int K, int Kc, float* __restrict__ partial, int N) {
    const int wave = threadIdx.x >> 5;
    const int lane = threadIdx.x & 31;
    const int half = lane >> 4;
    const int l16  = lane & 15;
    const int n0   = (blockIdx.x * 8 + wave) * 16;
    const int kstart = blockIdx.y * Kc;

    const float* wrow = W + (size_t)(n0 + l16) * K;
    const float* arow0 = A + (size_t)(0 * 16 + l16) * K;
    const float* arow1 = A + (size_t)(1 * 16 + l16) * K;
    const float* arow2 = A + (size_t)(2 * 16 + l16) * K;
    const float* arow3 = A + (size_t)(3 * 16 + l16) * K;

    floatx8 acc0 = {}, acc1 = {}, acc2 = {}, acc3 = {};

    const int kend = kstart + Kc;
    for (int k0 = kstart; k0 < kend; k0 += 32) {
        // prefetch the weight stream one tile ahead (global_prefetch_b8)
        __builtin_prefetch(wrow + k0 + 32, 0, 0);
        bf16x16 bfrag = cvt_load16(wrow + k0 + half * 8);
        bf16x16 a0 = cvt_loadA(arow0, k0, half);
        acc0 = __builtin_amdgcn_wmma_f32_16x16x32_bf16(false, a0, false, bfrag,
                                                       (short)0, acc0, false, false);
        bf16x16 a1 = cvt_loadA(arow1, k0, half);
        acc1 = __builtin_amdgcn_wmma_f32_16x16x32_bf16(false, a1, false, bfrag,
                                                       (short)0, acc1, false, false);
        bf16x16 a2 = cvt_loadA(arow2, k0, half);
        acc2 = __builtin_amdgcn_wmma_f32_16x16x32_bf16(false, a2, false, bfrag,
                                                       (short)0, acc2, false, false);
        bf16x16 a3 = cvt_loadA(arow3, k0, half);
        acc3 = __builtin_amdgcn_wmma_f32_16x16x32_bf16(false, a3, false, bfrag,
                                                       (short)0, acc3, false, false);
    }

    // C/D layout: element r -> M = r + half*8, N = l16. Store z as [n][b], b fast.
    float* out = partial + (size_t)blockIdx.y * N * 64;
    float* obase = out + (size_t)(n0 + l16) * 64 + half * 8;
#pragma unroll
    for (int m = 0; m < 4; ++m) {
        floatx8 a = (m == 0) ? acc0 : (m == 1) ? acc1 : (m == 2) ? acc2 : acc3;
        float* o = obase + m * 16;
        floatx4 lo = {a[0], a[1], a[2], a[3]};
        floatx4 hi = {a[4], a[5], a[6], a[7]};
        *(floatx4*)o       = lo;
        *(floatx4*)(o + 4) = hi;
    }
}

// z[i] = sum over S slices + bias1[n] (+ bias2[n]);  i over N*64, n = i/64
__global__ void combine_bias(const float* __restrict__ part, int S, int NB,
                             const float* __restrict__ b1, const float* __restrict__ b2,
                             float* __restrict__ z) {
    int i = blockIdx.x * 256 + threadIdx.x;
    if (i >= NB) return;
    float s = 0.f;
    for (int sl = 0; sl < S; ++sl) s += part[(size_t)sl * NB + i];
    int n = i >> 6;
    s += b1[n];
    if (b2) s += b2[n];
    z[i] = s;
}

// gates from z[(g*H+h)*64 + b]; c,out written [b*H+h]
__global__ void lstm_cell(const float* __restrict__ z, const float* __restrict__ cin,
                          float* __restrict__ out_h, float* __restrict__ out_c) {
    int idx = blockIdx.x * 256 + threadIdx.x;
    int b = idx & 63, h = idx >> 6;
    float zf = z[(size_t)(0 * HH + h) * 64 + b];
    float zi = z[(size_t)(1 * HH + h) * 64 + b];
    float zo = z[(size_t)(2 * HH + h) * 64 + b];
    float zg = z[(size_t)(3 * HH + h) * 64 + b];
    float c  = cin[(size_t)b * HH + h];
    float f  = 1.f / (1.f + __expf(-zf));
    float ii = 1.f / (1.f + __expf(-zi));
    float o  = 1.f / (1.f + __expf(-zo));
    float g  = tanhf(zg);
    float cn = f * c + ii * g;
    out_h[(size_t)b * HH + h] = o * tanhf(cn);
    out_c[(size_t)b * HH + h] = cn;
}

// per-batch-row online logsumexp of logits[n][b]*invT
__global__ void lse_kernel(const float* __restrict__ logits, float* __restrict__ lse) {
    int b = blockIdx.x, t = threadIdx.x;
    float m = -INFINITY, s = 0.f;
    for (int n = t; n < VV; n += 256) {
        float v = logits[(size_t)n * 64 + b] * INV_TEMP;
        if (v <= m) {
            s += __expf(v - m);
        } else {
            s = s * __expf(m - v) + 1.f;
            m = v;
        }
    }
    __shared__ float sm[256], ss[256];
    sm[t] = m; ss[t] = s;
    __syncthreads();
    for (int off = 128; off > 0; off >>= 1) {
        if (t < off) {
            float m2 = sm[t + off], s2 = ss[t + off];
            float M = fmaxf(sm[t], m2);
            ss[t] = ss[t] * __expf(sm[t] - M) + s2 * __expf(m2 - M);
            sm[t] = M;
        }
        __syncthreads();
    }
    if (t == 0) lse[b] = sm[0] + __logf(ss[0]);
}

__global__ void logp_kernel(const float* __restrict__ logits, const float* __restrict__ lse,
                            float* __restrict__ out) {
    int n = blockIdx.x * 256 + threadIdx.x;
    int b = blockIdx.y;
    out[(size_t)b * VV + n] = logits[(size_t)n * 64 + b] * INV_TEMP - lse[b];
}

extern "C" void kernel_launch(void* const* d_in, const int* in_sizes, int n_in,
                              void* d_out, int out_size, void* d_ws, size_t ws_size,
                              hipStream_t stream) {
    const float* x    = (const float*)d_in[0];
    const float* hs   = (const float*)d_in[1];   // [L,B,H]
    const float* cs   = (const float*)d_in[2];   // [L,B,H]
    const float* Wh   = (const float*)d_in[3];   // [L,4,H,H]
    const float* bh   = (const float*)d_in[4];   // [L,4,H]
    const float* Wx0  = (const float*)d_in[5];   // [4,H,V]
    const float* bx0  = (const float*)d_in[6];
    const float* Wx1  = (const float*)d_in[7];   // [4,H,H]
    const float* bx1  = (const float*)d_in[8];
    const float* Wdec = (const float*)d_in[9];   // [V,H]
    const float* bdec = (const float*)d_in[10];

    float* out  = (float*)d_out;
    float* logp = out;                                 // [B,V]
    float* outs = out + (size_t)BB * VV;               // [L,B,H]
    float* csts = outs + (size_t)LL * BB * HH;         // [L,B,H]

    float* ws   = (float*)d_ws;
    float* part = ws;                                  // 9 * 4096*64 floats
    const size_t partMax = (size_t)9 * N4H * 64;
    float* z    = ws + partMax;                        // max(4096*64, 32000*64)
    float* lse  = z + (size_t)VV * 64;

    // ---- layer 0:  z0 = x@Wx0^T (split-K S=8) + h0@Wh0^T + bx0 + bh0 ----
    gemm_partial<<<dim3(N4H / 128, 8), 256, 0, stream>>>(x, Wx0, VV, VV / 8, part, N4H);
    gemm_partial<<<dim3(N4H / 128, 1), 256, 0, stream>>>(hs, Wh, HH, HH,
                                                         part + (size_t)8 * N4H * 64, N4H);
    combine_bias<<<(N4H * 64) / 256, 256, 0, stream>>>(part, 9, N4H * 64, bx0, bh, z);
    lstm_cell<<<(BB * HH) / 256, 256, 0, stream>>>(z, cs, outs, csts);

    // ---- layer 1 ----
    gemm_partial<<<dim3(N4H / 128, 1), 256, 0, stream>>>(outs, Wx1, HH, HH, part, N4H);
    gemm_partial<<<dim3(N4H / 128, 1), 256, 0, stream>>>(hs + (size_t)BB * HH,
                                                         Wh + (size_t)4 * HH * HH, HH, HH,
                                                         part + (size_t)N4H * 64, N4H);
    combine_bias<<<(N4H * 64) / 256, 256, 0, stream>>>(part, 2, N4H * 64, bx1, bh + 4 * HH, z);
    lstm_cell<<<(BB * HH) / 256, 256, 0, stream>>>(z, cs + (size_t)BB * HH,
                                                   outs + (size_t)BB * HH,
                                                   csts + (size_t)BB * HH);

    // ---- decoder + log_softmax ----
    gemm_partial<<<dim3(VV / 128, 1), 256, 0, stream>>>(outs + (size_t)BB * HH, Wdec,
                                                        HH, HH, part, VV);
    combine_bias<<<(VV * 64) / 256, 256, 0, stream>>>(part, 1, VV * 64, bdec, nullptr, z);
    lse_kernel<<<BB, 256, 0, stream>>>(z, lse);
    logp_kernel<<<dim3(VV / 256, BB), 256, 0, stream>>>(z, lse, logp);
}
